// DownSampleWithSigma_52913997087002
// MI455X (gfx1250) — matrix operand order repaired and checked by reading, not compiled, not run on hardware
//
#include <hip/hip_runtime.h>
#include <hip/hip_bf16.h>
#include <math.h>

// ---- problem constants (from reference) ----
#define BB   8
#define CC   128
#define NN   4096
#define KN   32     // neighborhood
#define HH   4
#define DQ   32
#define MDS  2048

typedef __attribute__((ext_vector_type(16))) __bf16 v16bf;
typedef __attribute__((ext_vector_type(8)))  __bf16 v8bf;
typedef __attribute__((ext_vector_type(8)))  float  v8f;

// ---- output section offsets (elements), concatenated return order ----
//  pool_top (B,128,2048) | idx_top (B,4,2048) | pool_drop (B,128,2048) | idx_drop (B,4,2048)
#define OFF_PT ((size_t)0)
#define OFF_IT ((size_t)2097152)
#define OFF_PD ((size_t)2162688)
#define OFF_ID ((size_t)4259840)

// ============================================================
// 0a. tiled transpose x (B,C,N) f32 -> xbT (B,N,C) bf16
//     so WMMA K-dim fragments become contiguous vector loads
// ============================================================
__global__ void cvt_tr_kernel(const float* __restrict__ x, __bf16* __restrict__ xbT) {
    __shared__ __bf16 t[32][33];
    int tx = threadIdx.x, ty = threadIdx.y;          // (32, 8)
    int bid = blockIdx.x;
    int nt = bid & 127;  bid >>= 7;                  // N/32
    int ct = bid & 3;    bid >>= 2;                  // C/32
    int b  = bid;
    #pragma unroll
    for (int k = 0; k < 4; ++k) {
        int c = ct * 32 + ty + k * 8;
        int n = nt * 32 + tx;
        t[ty + k * 8][tx] = (__bf16)x[((size_t)b * CC + c) * NN + n];
    }
    __syncthreads();
    #pragma unroll
    for (int k = 0; k < 4; ++k) {
        int n = nt * 32 + ty + k * 8;
        int c = ct * 32 + tx;
        xbT[((size_t)b * NN + n) * CC + c] = t[tx][ty + k * 8];
    }
}

// ============================================================
// 0b. xx[b][n] = sum_c x^2  (f32 source for accuracy)
// ============================================================
__global__ void xx_kernel(const float* __restrict__ x, float* __restrict__ xx) {
    int tid = blockIdx.x * blockDim.x + threadIdx.x;   // B*N threads
    int b = tid >> 12, n = tid & (NN - 1);
    float s = 0.f;
    const float* xp = x + (size_t)b * CC * NN + n;
    #pragma unroll 8
    for (int c = 0; c < CC; ++c) s += xp[(size_t)c * NN] * xp[(size_t)c * NN];
    xx[(size_t)b * NN + n] = s;
}

// ---- fragment builders (CDNA5 ISA 7.12.2 bf16 layouts, wave32) ----
// A 16x32 (MxK): lane m = lane&15; K runs {8*half+0..7} and {16+8*half+0..7}
__device__ __forceinline__ v16bf load_afrag_bf16(const __bf16* __restrict__ row, int kc, int half8) {
    v8bf lo = *(const v8bf*)(row + kc + half8);
    v8bf hi = *(const v8bf*)(row + kc + 16 + half8);
    v16bf a;
    #pragma unroll
    for (int e = 0; e < 8; ++e) { a[e] = lo[e]; a[e + 8] = hi[e]; }
    return a;
}
__device__ __forceinline__ v16bf load_afrag_f32(const float* __restrict__ row, int kc, int half8) {
    v16bf a;
    #pragma unroll
    for (int e = 0; e < 8; ++e) {
        a[e]     = (__bf16)row[kc + half8 + e];
        a[e + 8] = (__bf16)row[kc + 16 + half8 + e];
    }
    return a;
}
// B 32x16 (KxN): lane n = lane&15; K runs {8*half+0..7} and {16+8*half+0..7}
__device__ __forceinline__ v16bf load_bfrag_bf16(const __bf16* __restrict__ row, int kc, int half16) {
    v8bf lo = *(const v8bf*)(row + kc + (half16 >> 1));
    v8bf hi = *(const v8bf*)(row + kc + 16 + (half16 >> 1));
    v16bf bm;
    #pragma unroll
    for (int e = 0; e < 8; ++e) { bm[e] = lo[e]; bm[e + 8] = hi[e]; }
    return bm;
}

// ============================================================
// 1. Q/K/V projections (WMMA): qkvT[mat][b][n][o] = (W_mat @ x)^T
//    one wave: W-tile fragments register-resident, sweep 8 column tiles
// ============================================================
__global__ void proj_kernel(const float* __restrict__ Wq, const float* __restrict__ Wk,
                            const float* __restrict__ Wv, const __bf16* __restrict__ xbT,
                            float* __restrict__ qkvT) {
    int lane = threadIdx.x & 31, wid = threadIdx.x >> 5;
    int g = blockIdx.x * 8 + wid;
    int ng  = g & 31;   g >>= 5;       // 32 groups of 128 columns
    int ot  = g & 7;    g >>= 3;       // 8 row tiles
    int b   = g & 7;    g >>= 3;       // batch
    int mat = g;                       // 0..2
    const float* W = (mat == 0) ? Wq : (mat == 1) ? Wk : Wv;
    int o0 = ot * 16;
    int half8  = (lane >> 4) << 3;
    int half16 = (lane >> 4) << 4;

    // A fragments (weights), 4 K-chunks, kept in registers
    const float* wrow = W + (size_t)(o0 + (lane & 15)) * CC;
    v16bf afrag[4];
    #pragma unroll
    for (int kc = 0; kc < 4; ++kc) afrag[kc] = load_afrag_f32(wrow, kc * 32, half8);

    float* OT = qkvT + ((size_t)mat * BB + b) * NN * CC;
    const __bf16* XT = xbT + (size_t)b * NN * CC;

    #pragma unroll
    for (int t = 0; t < 8; ++t) {
        int n0 = ng * 128 + t * 16;
        const __bf16* brow = XT + (size_t)(n0 + (lane & 15)) * CC;
        v8f acc = {};
        #pragma unroll
        for (int kc = 0; kc < 4; ++kc) {
            v16bf bm = load_bfrag_bf16(brow, kc * 32, half16);
            acc = __builtin_amdgcn_wmma_f32_16x16x32_bf16(false, afrag[kc], false, bm,
                                                          (short)0, acc, false, false);
        }
        // D element r: row o0+r+8*half, col n0+(lane&15); store transposed (n-major, o contiguous)
        float* dst = OT + (size_t)(n0 + (lane & 15)) * CC + o0 + half8;
        *(float4*)(dst)     = make_float4(acc[0], acc[1], acc[2], acc[3]);
        *(float4*)(dst + 4) = make_float4(acc[4], acc[5], acc[6], acc[7]);
    }
}

// ============================================================
// 2. Pairwise distance (WMMA) fused with per-row top-32
//    one wave per (b, 16-row tile); A fragments hoisted out of column loop
// ============================================================
__global__ void dist_topk_kernel(const __bf16* __restrict__ xbT, const float* __restrict__ xx,
                                 int* __restrict__ knn) {
    __shared__ float tile[16][17];
    __shared__ float lv[16][33];
    __shared__ int   li[16][33];
    int lane = threadIdx.x;
    int id = blockIdx.x;
    int mt = id & 255, b = id >> 8;
    int m0 = mt * 16;
    int half8  = (lane >> 4) << 3;
    int half16 = (lane >> 4) << 4;

    for (int i = lane; i < 16 * 32; i += 32) { lv[i >> 5][i & 31] = 3.4e38f; li[i >> 5][i & 31] = 0; }
    __syncthreads();
    float curMax = 3.4e38f;

    const __bf16* XT  = xbT + (size_t)b * NN * CC;
    const float*  xxb = xx + (size_t)b * NN;

    // A fragments: rows m0..m0+15 of X^T, invariant across all column tiles
    const __bf16* arow = XT + (size_t)(m0 + (lane & 15)) * CC;
    v16bf afrag[4];
    #pragma unroll
    for (int kc = 0; kc < 4; ++kc) afrag[kc] = load_afrag_bf16(arow, kc * 32, half8);

    float xxm[8];
    #pragma unroll
    for (int r = 0; r < 8; ++r) xxm[r] = xxb[m0 + r + half8];

    for (int jt = 0; jt < NN / 16; ++jt) {
        int j0 = jt * 16;
        const __bf16* brow = XT + (size_t)(j0 + (lane & 15)) * CC;
        __builtin_prefetch(XT + (size_t)(j0 + 16 + (lane & 15)) * CC, 0, 1);
        v8f acc = {};
        #pragma unroll
        for (int kc = 0; kc < 4; ++kc) {
            v16bf bm = load_bfrag_bf16(brow, kc * 32, half16);
            acc = __builtin_amdgcn_wmma_f32_16x16x32_bf16(false, afrag[kc], false, bm,
                                                          (short)0, acc, false, false);
        }
        float xxj = xxb[j0 + (lane & 15)];
        #pragma unroll
        for (int r = 0; r < 8; ++r)
            tile[r + half8][lane & 15] = xxm[r] - 2.0f * acc[r] + xxj;
        __syncthreads();
        if (lane < 16) {
            for (int j = 0; j < 16; ++j) {
                float d = tile[lane][j];
                if (d < curMax) {
                    int p = 31;
                    while (p > 0 && lv[lane][p - 1] > d) {
                        lv[lane][p] = lv[lane][p - 1];
                        li[lane][p] = li[lane][p - 1];
                        --p;
                    }
                    lv[lane][p] = d; li[lane][p] = j0 + j;
                    curMax = lv[lane][31];
                }
            }
        }
        __syncthreads();
    }
    if (lane < 16) {
        int* kp = knn + ((size_t)b * NN + m0 + lane) * KN;
        for (int j = 0; j < KN; ++j) kp[j] = li[lane][j];
    }
}

// ============================================================
// 3. energy -> softmax -> std  (one wave per (b,h,n); lane = neighbor)
//    transposed Q/K: per-lane rows are contiguous 128B reads
// ============================================================
__global__ void attn_kernel(const float* __restrict__ qkvT, const int* __restrict__ knn,
                            float* __restrict__ attn, float* __restrict__ score) {
    int lane = threadIdx.x & 31, wid = threadIdx.x >> 5;
    int g = blockIdx.x * 8 + wid;
    int n = g & (NN - 1);  g >>= 12;
    int h = g & (HH - 1);  g >>= 2;
    int b = g;
    const float* QT = qkvT;
    const float* KT = qkvT + (size_t)BB * NN * CC;
    int nb = knn[((size_t)b * NN + n) * KN + lane];

    const float4* q4 = (const float4*)(QT + ((size_t)b * NN + n)  * CC + h * DQ);
    const float4* a4 = (const float4*)(KT + ((size_t)b * NN + nb) * CC + h * DQ);
    const float4* c4 = (const float4*)(KT + ((size_t)b * NN + n)  * CC + h * DQ);
    float e = 0.f;
    #pragma unroll
    for (int i = 0; i < DQ / 4; ++i) {
        float4 q = q4[i], ka = a4[i], kc = c4[i];
        e += q.x * (ka.x - kc.x) + q.y * (ka.y - kc.y)
           + q.z * (ka.z - kc.z) + q.w * (ka.w - kc.w);
    }
    e *= 0.17677669529663687f;   // 1/sqrt(32)

    float mx = e;
    #pragma unroll
    for (int off = 16; off; off >>= 1) mx = fmaxf(mx, __shfl_xor(mx, off, 32));
    float p = __expf(e - mx);
    float s = p;
    #pragma unroll
    for (int off = 16; off; off >>= 1) s += __shfl_xor(s, off, 32);
    float a = p / s;
    float ss = a * a;
    #pragma unroll
    for (int off = 16; off; off >>= 1) ss += __shfl_xor(ss, off, 32);

    attn[(((size_t)(b * HH + h) * NN) + n) * KN + lane] = a;
    if (lane == 0) {
        const float mean = 1.0f / 32.0f;            // softmax sums to 1
        float var = ss * (1.0f / 32.0f) - mean * mean;
        score[(size_t)(b * HH + h) * NN + n] = sqrtf(fmaxf(var, 0.f));
    }
}

// ============================================================
// 4. per-(b,h) bitonic sort of 4096 (score,idx); emit idx_top / idx_drop
// ============================================================
__global__ void sort_kernel(const float* __restrict__ score, int* __restrict__ sel,
                            int* __restrict__ iout) {
    __shared__ float sv[NN];
    __shared__ int   si[NN];
    int t = threadIdx.x;                 // 1024 threads
    int bh = blockIdx.x;                 // 0..31
    const float* sc = score + (size_t)bh * NN;
    for (int i = t; i < NN; i += 1024) { sv[i] = sc[i]; si[i] = i; }
    __syncthreads();

    for (int ksz = 2; ksz <= NN; ksz <<= 1) {
        for (int jsz = ksz >> 1; jsz > 0; jsz >>= 1) {
            for (int i = t; i < NN / 2; i += 1024) {
                int idx = ((i / jsz) * (jsz << 1)) + (i % jsz);
                int ixj = idx + jsz;
                bool desc = ((idx & ksz) == 0);      // overall descending
                float a0 = sv[idx], a1 = sv[ixj];
                bool dosw = desc ? (a0 < a1) : (a0 > a1);
                if (dosw) {
                    sv[idx] = a1; sv[ixj] = a0;
                    int tmp = si[idx]; si[idx] = si[ixj]; si[ixj] = tmp;
                }
            }
            __syncthreads();
        }
    }
    for (int i = t; i < MDS; i += 1024) {
        int topIdx  = si[i];             // descending score -> top_k order
        int dropIdx = si[NN - 1 - i];    // ascending score  -> top_k(-score) order
        sel[(size_t)bh * NN + i]        = topIdx;
        sel[(size_t)bh * NN + MDS + i]  = dropIdx;
        iout[OFF_IT + (size_t)bh * MDS + i] = topIdx;
        iout[OFF_ID + (size_t)bh * MDS + i] = dropIdx;
    }
}

// ============================================================
// 5. pooling: o = sum_j attn_j * VT[nbr_j][:] - VT[n][:]  (sum attn = 1)
//    one wave per (sec,b,h,m); lane = feature d -> 128B coalesced per neighbor
// ============================================================
__global__ void pool_kernel(const float* __restrict__ qkvT, const float* __restrict__ attn,
                            const int* __restrict__ knn, const int* __restrict__ sel,
                            float* __restrict__ out) {
    int lane = threadIdx.x & 31, wid = threadIdx.x >> 5;
    int g = blockIdx.x * 8 + wid;
    int m = g & (MDS - 1); g >>= 11;
    int h = g & (HH - 1);  g >>= 2;
    int b = g & 7;         g >>= 3;
    int sec = g;                        // 0 = top, 1 = drop
    int bh = b * HH + h;
    int p = sel[(size_t)bh * NN + sec * MDS + m];

    const float* VT   = qkvT + (size_t)2 * BB * NN * CC + (size_t)b * NN * CC;
    const float* arow = attn + ((size_t)bh * NN + p) * KN;
    const int*   krow = knn + ((size_t)b * NN + p) * KN;

    float acc = 0.f;
    #pragma unroll 8
    for (int j = 0; j < KN; ++j) {
        int nb = krow[j];
        acc += arow[j] * VT[(size_t)nb * CC + h * DQ + lane];
    }
    acc -= VT[(size_t)p * CC + h * DQ + lane];

    size_t base = sec ? OFF_PD : OFF_PT;
    out[base + ((size_t)b * CC + h * DQ + lane) * MDS + m] = acc;
}

// ============================================================
extern "C" void kernel_launch(void* const* d_in, const int* in_sizes, int n_in,
                              void* d_out, int out_size, void* d_ws, size_t ws_size,
                              hipStream_t stream) {
    (void)in_sizes; (void)n_in; (void)out_size; (void)ws_size;
    const float* x  = (const float*)d_in[0];
    const float* Wq = (const float*)d_in[1];
    const float* Wk = (const float*)d_in[2];
    const float* Wv = (const float*)d_in[3];
    float* out = (float*)d_out;
    int*   iout = (int*)d_out;

    // ---- workspace layout (bytes) ----
    char* ws = (char*)d_ws;
    __bf16* xbT  = (__bf16*)(ws);                              //  8 MB  (B,N,C) bf16
    float*  xx   = (float*)(ws + ((size_t)8  << 20));          //  128 KB (pad 1MB)
    float*  qkvT = (float*)(ws + ((size_t)9  << 20));          //  48 MB (Q,K,V transposed (B,N,C))
    int*    knn  = (int*)  (ws + ((size_t)57 << 20));          //  4 MB
    float*  attn = (float*)(ws + ((size_t)61 << 20));          //  16 MB
    float*  score= (float*)(ws + ((size_t)77 << 20));          //  0.5 MB (pad 1MB)
    int*    sel  = (int*)  (ws + ((size_t)78 << 20));          //  0.5 MB

    // 0. bf16 transpose of x + row norms
    cvt_tr_kernel<<<BB * 4 * 128, dim3(32, 8), 0, stream>>>(x, xbT);
    xx_kernel<<<(BB * NN) / 256, 256, 0, stream>>>(x, xx);

    // 1. Q/K/V projections (WMMA): 3*8*8*32 waves / 8 per block
    proj_kernel<<<768, 256, 0, stream>>>(Wq, Wk, Wv, xbT, qkvT);

    // 2. distance GEMM (WMMA) + fused per-row top-32
    dist_topk_kernel<<<BB * (NN / 16), 32, 0, stream>>>(xbT, xx, knn);

    // 3. energy/softmax/std
    attn_kernel<<<(BB * HH * NN) / 8, 256, 0, stream>>>(qkvT, knn, attn, score);

    // 4. per-(b,h) sort -> idx_top / idx_drop
    sort_kernel<<<BB * HH, 1024, 0, stream>>>(score, sel, iout);

    // 5. pooling for both sections
    pool_kernel<<<(2 * BB * HH * MDS) / 8, 256, 0, stream>>>(qkvT, attn, knn, sel, out);
}
